// Attention_47029891891914
// MI455X (gfx1250) — matrix operand order repaired and checked by reading, not compiled, not run on hardware
//
#include <hip/hip_runtime.h>

// ---------------------------------------------------------------------------
// MHA block for MI455X (gfx1250): bf16 WMMA everywhere + TDM-staged attention.
//   DIM=1024, HEADS=16, HD=64, B=8, N=1024  -> M = B*N = 8192 tokens
// ---------------------------------------------------------------------------

typedef __attribute__((ext_vector_type(16))) __bf16 v16bf;
typedef __attribute__((ext_vector_type(8)))  __bf16 v8bf;
typedef __attribute__((ext_vector_type(8)))  float  v8f;

typedef __attribute__((ext_vector_type(4))) unsigned tdm_g0_t;  // D# group 0
typedef __attribute__((ext_vector_type(8))) int      tdm_g1_t;  // D# group 1
typedef __attribute__((ext_vector_type(4))) int      tdm_g2_t;  // D# groups 2/3

#define DIM      1024
#define NHEADS   16
#define HD       64
#define BATCH    8
#define SEQ      1024
#define MTOT     (BATCH * SEQ)        // 8192
#define SCALE_F  0.125f               // 64^-0.5
#define BHSZ     (SEQ * HD)           // 65536 elems per (b,h) plane

static __device__ inline __bf16 f2b(float f) {
    unsigned u = __builtin_bit_cast(unsigned, f);
    unsigned r = (u + 0x7FFFu + ((u >> 16) & 1u)) >> 16;
    return __builtin_bit_cast(__bf16, (unsigned short)r);
}

static __device__ inline v8f wmma_bf16(v16bf a, v16bf b, v8f c) {
    return __builtin_amdgcn_wmma_f32_16x16x32_bf16(
        false, a, false, b, (short)0, c, false, false);
}

// Load a 16-element bf16 fragment as two 16B chunks, `gap` elements apart.
// gap=16 -> A-fragment chunking (K = 8h..8h+7, 16+8h..16+8h+7)
// gap=8  -> contiguous K = 16h..16h+15, used for both A and B here
static __device__ inline v16bf load_frag(const __bf16* p, int gap) {
    v8bf lo = *(const v8bf*)p;
    v8bf hi = *(const v8bf*)(p + gap);
    return __builtin_shufflevector(lo, hi, 0,1,2,3,4,5,6,7,8,9,10,11,12,13,14,15);
}

// ---------------------------------------------------------------------------
// TDM: issue a 2D tile load (bf16 elements) into LDS.
// D# bitfields per CDNA5 ISA:
//   g0: [1:0]=count(1) | [63:32]=lds_addr | [120:64]=global_addr | [127:126]=2
//   g1: [17:16]=data_size(1 -> 2B), [79:48]=tensor_dim0, [111:80]=tensor_dim1,
//       [127:112]=tile_dim0, [143:128]=tile_dim1, [207:160]=tensor_dim0_stride
// Dims/strides in element (data_size) units; addresses in bytes.
// This toolchain's builtin is the 6-arg form:
//   (uint32x4 g0, int32x8 g1, int32x4 g2, int32x4 g3, int32x8 extra, i32 cpol)
// ---------------------------------------------------------------------------
static __device__ inline void tdm_load_2d(unsigned lds_byte_off,
                                          const __bf16* gptr,
                                          unsigned tensor_d0, unsigned tensor_d1,
                                          unsigned tile_d0,   unsigned tile_d1,
                                          unsigned stride0) {
    unsigned long long ga = (unsigned long long)gptr;
    tdm_g0_t g0;
    g0[0] = 1u;                                              // count=1 (user mode)
    g0[1] = lds_byte_off;                                    // lds_addr
    g0[2] = (unsigned)(ga & 0xFFFFFFFFu);                    // global_addr[31:0]
    g0[3] = (unsigned)((ga >> 32) & 0x01FFFFFFu) | (2u << 30); // addr[56:32] | type=2
    tdm_g1_t g1;
    g1[0] = (int)(1u << 16);                                 // data_size=1(2B), wg_mask=0
    g1[1] = (int)((tensor_d0 & 0xFFFFu) << 16);              // tensor_dim0[15:0]
    g1[2] = (int)(((tensor_d0 >> 16) & 0xFFFFu) | ((tensor_d1 & 0xFFFFu) << 16));
    g1[3] = (int)(((tensor_d1 >> 16) & 0xFFFFu) | ((tile_d0 & 0xFFFFu) << 16));
    g1[4] = (int)(tile_d1 & 0xFFFFu);                        // tile_dim2 = 0
    g1[5] = (int)stride0;                                    // dim0_stride[31:0]
    g1[6] = 0;                                               // stride hi / dim1_stride
    g1[7] = 0;
    tdm_g2_t z4 = {0, 0, 0, 0};                              // 2D: groups 2/3 unused
    tdm_g1_t z8 = {0, 0, 0, 0, 0, 0, 0, 0};
    __builtin_amdgcn_tensor_load_to_lds(g0, g1, z4, z4, z8, 0);
}

// ---------------------------------------------------------------------------
// Kernel 1: fp32 -> bf16 elementwise (activations)
// ---------------------------------------------------------------------------
__global__ void k_cvt_bf16(const float* __restrict__ x, __bf16* __restrict__ y, long n) {
    long i = ((long)blockIdx.x * blockDim.x + threadIdx.x) * 4;
    if (i + 3 < n) {
        float4 v = *(const float4*)(x + i);
        y[i + 0] = f2b(v.x);
        y[i + 1] = f2b(v.y);
        y[i + 2] = f2b(v.z);
        y[i + 3] = f2b(v.w);
    }
}

// ---------------------------------------------------------------------------
// Kernel 2: LDS-tiled transpose + convert: w[K,N] fp32 -> wT[N,K] bf16
// Launch: grid(N/32, K/32), block(32,8).  K,N multiples of 32.
// ---------------------------------------------------------------------------
__global__ void k_transpose_cvt(const float* __restrict__ w, __bf16* __restrict__ wT,
                                int K, int N) {
    __shared__ float tile[32][33];
    int n = blockIdx.x * 32 + threadIdx.x;
    int k = blockIdx.y * 32 + threadIdx.y;
    #pragma unroll
    for (int j = 0; j < 32; j += 8)
        tile[threadIdx.y + j][threadIdx.x] = w[(size_t)(k + j) * N + n];
    __syncthreads();
    int n2 = blockIdx.x * 32 + threadIdx.y;
    int k2 = blockIdx.y * 32 + threadIdx.x;
    #pragma unroll
    for (int j = 0; j < 32; j += 8)
        wT[(size_t)(n2 + j) * K + k2] = f2b(tile[threadIdx.x][threadIdx.y + j]);
}

// ---------------------------------------------------------------------------
// Kernel 3: QKV GEMM.  A = inp_bf16 [8192,1024], WT = w_qkvT [3072,1024].
// One wave -> 32x64 output tile (8 WMMAs per K=32 step).
// Epilogue splits into q [bh,n,64], k [bh,n,64], vT [bh,64,n]  (bf16).
// ---------------------------------------------------------------------------
__global__ void k_qkv_gemm(const __bf16* __restrict__ A,
                           const __bf16* __restrict__ WT,
                           const float*  __restrict__ bias,
                           __bf16* __restrict__ qo,
                           __bf16* __restrict__ ko,
                           __bf16* __restrict__ vTo) {
    const int wave = (blockIdx.x * blockDim.x + threadIdx.x) >> 5;
    const int lane = threadIdx.x & 31;
    const int h16  = lane >> 4;
    const int l16  = lane & 15;

    const int ntile = wave % 48;           // 48 tiles of 64 along N=3072
    const int mtile = wave / 48;           // 256 tiles of 32 along M=8192
    const int m0 = mtile * 32;
    const int n0 = ntile * 64;

    v8f acc[2][4] = {};

    for (int k0 = 0; k0 < DIM; k0 += 32) {
        v16bf a[2];
        #pragma unroll
        for (int mi = 0; mi < 2; ++mi) {
            const __bf16* p = A + (size_t)(m0 + mi * 16 + l16) * DIM + k0 + h16 * 16;
            a[mi] = load_frag(p, 8);
        }
        #pragma unroll
        for (int nj = 0; nj < 4; ++nj) {
            const __bf16* p = WT + (size_t)(n0 + nj * 16 + l16) * DIM + k0 + h16 * 16;
            v16bf b = load_frag(p, 8);
            acc[0][nj] = wmma_bf16(a[0], b, acc[0][nj]);
            acc[1][nj] = wmma_bf16(a[1], b, acc[1][nj]);
        }
    }

    const int sec = ntile / 16;            // 0=q, 1=k, 2=v
    const int hh  = ntile % 16;            // head index
    #pragma unroll
    for (int mi = 0; mi < 2; ++mi) {
        #pragma unroll
        for (int nj = 0; nj < 4; ++nj) {
            const int n  = n0 + nj * 16 + l16;
            const float bv = bias[n];
            const int d  = nj * 16 + l16;  // 0..63 within head
            #pragma unroll
            for (int r = 0; r < 8; ++r) {
                const int m  = m0 + mi * 16 + r + 8 * h16;
                const int bb = m >> 10;
                const int t  = m & 1023;
                const __bf16 o = f2b(acc[mi][nj][r] + bv);
                const size_t bh = (size_t)(bb * NHEADS + hh);
                if (sec == 0)      qo [bh * BHSZ + (size_t)t * HD + d]  = o;
                else if (sec == 1) ko [bh * BHSZ + (size_t)t * HD + d]  = o;
                else               vTo[bh * BHSZ + (size_t)d * SEQ + t] = o;
            }
        }
    }
}

// ---------------------------------------------------------------------------
// Kernel 4: flash attention with TDM-staged K/V tiles in LDS.
// Block = 8 waves, all on one (b,h); wave w owns 16 queries (128 per block).
// Per 128-key stage, wave 0 DMAs K[128,64] and vT[64,128] tiles into LDS via
// tensor_load_to_lds (double buffered, TENSORcnt-gated), all waves consume
// via ds_load_b128 fragments.
//   S^T = K_tile . Q^T  -> per-lane softmax;  P^T B-frag via 16 shfl_xor(16);
//   O^T = V^T . P^T accumulated online.  8 WMMAs per 32 keys.
// ---------------------------------------------------------------------------
struct AttnLds {
    __bf16 kbuf[2][128][HD];   // 16 KB per buffer  (LDS offsets 0, 16384)
    __bf16 vbuf[2][HD][128];   // 16 KB per buffer  (LDS offsets 32768, 49152)
};
#define KOFF(b) ((unsigned)((b) * 16384))
#define VOFF(b) ((unsigned)(32768 + (b) * 16384))

__global__ void k_attention(const __bf16* __restrict__ q,
                            const __bf16* __restrict__ k,
                            const __bf16* __restrict__ vT,
                            __bf16* __restrict__ out) {
    __shared__ AttnLds sm;

    const int tid  = threadIdx.x;
    const int widx = tid >> 5;
    const int lane = tid & 31;
    const int h16  = lane >> 4;
    const int l16  = lane & 15;

    const int bh   = blockIdx.x >> 3;      // 0..127
    const int qblk = blockIdx.x & 7;       // 8 query blocks of 128
    const int q0   = qblk * 128 + widx * 16;

    // The TDM engine writes `sm` behind the compiler's back; plant an opaque
    // never-taken store (bh is unbounded at compile time, indices dynamic) so
    // the optimizer cannot fold the LDS fragment loads of never-written
    // memory to undef.
    if (bh > 100000) {
        sm.kbuf[bh & 1][tid & 127][tid & 63]  = (__bf16)0.0f;
        sm.vbuf[bh & 1][tid & 63][tid & 127]  = (__bf16)0.0f;
    }

    const __bf16* qp = q  + (size_t)bh * BHSZ;
    const __bf16* kp = k  + (size_t)bh * BHSZ;
    const __bf16* vp = vT + (size_t)bh * BHSZ;

    // Q^T B-fragments (hd split 0..31 / 32..63), loaded once from global.
    v16bf qb[2];
    #pragma unroll
    for (int j = 0; j < 2; ++j) {
        const __bf16* p = qp + (size_t)(q0 + l16) * HD + j * 32 + h16 * 16;
        qb[j] = load_frag(p, 8);
    }

    v8f   o[4] = {};
    float m = -3.0e38f;
    float l = 0.0f;

    const int NSTAGE = SEQ / 128;          // 8 stages

    // Prologue: stage 0 DMA (wave 0 issues both descriptors).
    if (widx == 0) {
        tdm_load_2d(KOFF(0), kp, /*t_d0=*/HD, /*t_d1=*/SEQ, /*tile=*/HD, 128, /*stride0=*/HD);
        tdm_load_2d(VOFF(0), vp, /*t_d0=*/SEQ, /*t_d1=*/HD, /*tile=*/128, HD, /*stride0=*/SEQ);
    }

    for (int s = 0; s < NSTAGE; ++s) {
        const int buf = s & 1;
        if (widx == 0) __builtin_amdgcn_s_wait_tensorcnt(0);
        __syncthreads();                   // stage data visible; prev buffer free
        if (widx == 0 && s + 1 < NSTAGE) {
            const int kk2 = (s + 1) * 128;
            tdm_load_2d(KOFF(buf ^ 1), kp + (size_t)kk2 * HD, HD, SEQ, HD, 128, HD);
            tdm_load_2d(VOFF(buf ^ 1), vp + kk2,              SEQ, HD, 128, HD, SEQ);
        }

        #pragma unroll
        for (int kb = 0; kb < 4; ++kb) {   // 4 x 32 keys per stage
            // --- S^T tiles (16 keys x 16 queries) from LDS K tile
            v8f sacc[2] = {};
            #pragma unroll
            for (int t = 0; t < 2; ++t) {
                #pragma unroll
                for (int ks = 0; ks < 2; ++ks) {
                    const __bf16* p = &sm.kbuf[buf][kb * 32 + t * 16 + l16][ks * 32 + h16 * 16];
                    v16bf a = load_frag(p, 8);
                    sacc[t] = wmma_bf16(a, qb[ks], sacc[t]);
                }
            }

            // --- online softmax (per-lane over this lane's 16 keys)
            float sv0[8], sv1[8];
            float mx = -3.0e38f;
            #pragma unroll
            for (int r = 0; r < 8; ++r) {
                sv0[r] = sacc[0][r] * SCALE_F;
                sv1[r] = sacc[1][r] * SCALE_F;
                mx = fmaxf(mx, fmaxf(sv0[r], sv1[r]));
            }
            mx = fmaxf(mx, __shfl_xor(mx, 16, 32));
            const float mnew  = fmaxf(m, mx);
            const float alpha = __expf(m - mnew);
            m = mnew;

            float ps = 0.0f;
            #pragma unroll
            for (int r = 0; r < 8; ++r) {
                sv0[r] = __expf(sv0[r] - mnew);
                sv1[r] = __expf(sv1[r] - mnew);
                ps += sv0[r] + sv1[r];
            }
            l = l * alpha + ps;

            #pragma unroll
            for (int dt = 0; dt < 4; ++dt)
                #pragma unroll
                for (int r = 0; r < 8; ++r)
                    o[dt][r] *= alpha;

            // --- P^T B-fragment: element e holds key (base + 16*h16 + e)
            v16bf pb;
            #pragma unroll
            for (int r = 0; r < 8; ++r) {
                const float x0 = __shfl_xor(sv0[r], 16, 32);
                const float x1 = __shfl_xor(sv1[r], 16, 32);
                pb[r]     = f2b(h16 ? x1     : sv0[r]);
                pb[r + 8] = f2b(h16 ? sv1[r] : x0);
            }

            // --- O^T += V^T_tile . P^T   (4 d-blocks of 16) from LDS V tile
            #pragma unroll
            for (int dt = 0; dt < 4; ++dt) {
                const __bf16* p = &sm.vbuf[buf][dt * 16 + l16][kb * 32 + h16 * 16];
                v16bf a = load_frag(p, 8);
                o[dt] = wmma_bf16(a, pb, o[dt]);
            }
        }
        __syncthreads();                   // all waves done with buf this stage
    }

    const float lt  = l + __shfl_xor(l, 16, 32);
    const float inv = 1.0f / lt;

    // out layout: [token = b*1024+query, dim = h*64+d]  (bf16, A-frag-ready)
    const int bb = bh >> 4;
    const int hh = bh & 15;
    const size_t row = (size_t)(bb * SEQ + q0 + l16);
    #pragma unroll
    for (int dt = 0; dt < 4; ++dt) {
        #pragma unroll
        for (int r = 0; r < 8; ++r) {
            const int d = dt * 16 + r + 8 * h16;
            out[row * DIM + hh * HD + d] = f2b(o[dt][r] * inv);
        }
    }
}

// ---------------------------------------------------------------------------
// Kernel 5: output projection.  A = attn_bf16 [8192,1024], WT = w_projT
// [1024,1024].  32x64 per wave, fp32 epilogue + bias.
// ---------------------------------------------------------------------------
__global__ void k_proj_gemm(const __bf16* __restrict__ A,
                            const __bf16* __restrict__ WT,
                            const float*  __restrict__ bias,
                            float* __restrict__ out) {
    const int wave = (blockIdx.x * blockDim.x + threadIdx.x) >> 5;
    const int lane = threadIdx.x & 31;
    const int h16  = lane >> 4;
    const int l16  = lane & 15;

    const int ntile = wave % 16;           // 16 tiles of 64 along N=1024
    const int mtile = wave / 16;           // 256 tiles of 32 along M=8192
    const int m0 = mtile * 32;
    const int n0 = ntile * 64;

    v8f acc[2][4] = {};

    for (int k0 = 0; k0 < DIM; k0 += 32) {
        v16bf a[2];
        #pragma unroll
        for (int mi = 0; mi < 2; ++mi) {
            const __bf16* p = A + (size_t)(m0 + mi * 16 + l16) * DIM + k0 + h16 * 16;
            a[mi] = load_frag(p, 8);
        }
        #pragma unroll
        for (int nj = 0; nj < 4; ++nj) {
            const __bf16* p = WT + (size_t)(n0 + nj * 16 + l16) * DIM + k0 + h16 * 16;
            v16bf b = load_frag(p, 8);
            acc[0][nj] = wmma_bf16(a[0], b, acc[0][nj]);
            acc[1][nj] = wmma_bf16(a[1], b, acc[1][nj]);
        }
    }

    #pragma unroll
    for (int mi = 0; mi < 2; ++mi) {
        #pragma unroll
        for (int nj = 0; nj < 4; ++nj) {
            const int n = n0 + nj * 16 + l16;
            const float bv = bias[n];
            #pragma unroll
            for (int r = 0; r < 8; ++r) {
                const int mrow = m0 + mi * 16 + r + 8 * h16;
                out[(size_t)mrow * DIM + n] = acc[mi][nj][r] + bv;
            }
        }
    }
}

// ---------------------------------------------------------------------------
// Host launcher
// ---------------------------------------------------------------------------
extern "C" void kernel_launch(void* const* d_in, const int* in_sizes, int n_in,
                              void* d_out, int out_size, void* d_ws, size_t ws_size,
                              hipStream_t stream) {
    const float* inp    = (const float*)d_in[0];   // [8,1024,1024]
    const float* w_qkv  = (const float*)d_in[1];   // [1024,3072]
    const float* b_qkv  = (const float*)d_in[2];   // [3072]
    const float* w_proj = (const float*)d_in[3];   // [1024,1024]
    const float* b_proj = (const float*)d_in[4];   // [1024]
    float* out = (float*)d_out;

    char* ws = (char*)d_ws;
    const size_t SZ_INP   = (size_t)MTOT * DIM * 2;        // 16 MB
    const size_t SZ_WQKV  = (size_t)3 * DIM * DIM * 2;     //  6 MB
    const size_t SZ_WPROJ = (size_t)DIM * DIM * 2;         //  2 MB
    const size_t SZ_PLANE = (size_t)MTOT * DIM * 2;        // 16 MB each (q,k,vT,attn)

    __bf16* inp_b   = (__bf16*)(ws);
    __bf16* wqkvT   = (__bf16*)(ws + SZ_INP);
    __bf16* wprojT  = (__bf16*)(ws + SZ_INP + SZ_WQKV);
    __bf16* qbuf    = (__bf16*)(ws + SZ_INP + SZ_WQKV + SZ_WPROJ);
    __bf16* kbuf    = (__bf16*)(ws + SZ_INP + SZ_WQKV + SZ_WPROJ + SZ_PLANE);
    __bf16* vTbuf   = (__bf16*)(ws + SZ_INP + SZ_WQKV + SZ_WPROJ + 2 * SZ_PLANE);
    __bf16* attnbuf = (__bf16*)(ws + SZ_INP + SZ_WQKV + SZ_WPROJ + 3 * SZ_PLANE);

    // 1) convert activations to bf16
    {
        const long n = (long)MTOT * DIM;                    // 8388608
        const int  threads = 256;
        const int  blocks  = (int)(n / 4 / threads);        // 8192
        k_cvt_bf16<<<blocks, threads, 0, stream>>>(inp, inp_b, n);
    }
    // 2) transpose+convert weights
    {
        dim3 blk(32, 8);
        k_transpose_cvt<<<dim3(3 * DIM / 32, DIM / 32), blk, 0, stream>>>(w_qkv, wqkvT, DIM, 3 * DIM);
        k_transpose_cvt<<<dim3(DIM / 32, DIM / 32),     blk, 0, stream>>>(w_proj, wprojT, DIM, DIM);
    }
    // 3) QKV GEMM: 256 m-tiles x 48 n-tiles = 12288 waves -> 1536 blocks x 8 waves
    k_qkv_gemm<<<1536, 256, 0, stream>>>(inp_b, wqkvT, b_qkv, qbuf, kbuf, vTbuf);
    // 4) attention: 128 (b,h) x 8 query-blocks = 1024 blocks x 8 waves
    k_attention<<<1024, 256, 0, stream>>>(qbuf, kbuf, vTbuf, attnbuf);
    // 5) projection: 256 m-tiles x 16 n-tiles = 4096 waves -> 512 blocks x 8 waves
    k_proj_gemm<<<512, 256, 0, stream>>>(attnbuf, wprojT, b_proj, out);

    (void)in_sizes; (void)n_in; (void)out_size; (void)ws_size;
}
